// MultiDimensionalRNN_20727512170836
// MI455X (gfx1250) — matrix operand order repaired and checked by reading, not compiled, not run on hardware
//
#include <hip/hip_runtime.h>

#define B_ 128
#define R_ 64
#define C_ 64
#define F_ 128
#define U_ 128
#define WPAD 136                          // padded LDS row stride (bf16 elems): bank-conflict-free
#define LDS_BYTES (4 * U_ * WPAD * sizeof(__bf16))   // 139264 B < 320 KB/WGP

typedef __attribute__((ext_vector_type(16))) __bf16 v16bf;
typedef __attribute__((ext_vector_type(8)))  __bf16 v8bf;
typedef __attribute__((ext_vector_type(8)))  float  v8f;
typedef __attribute__((ext_vector_type(4))) unsigned int u32x4;
typedef __attribute__((ext_vector_type(8))) int i32x8;
typedef __attribute__((ext_vector_type(4))) int i32x4;

// ---------------------------------------------------------------------------
// Prep: convert W[F][U], Ub/Uu/Ud[U][U] (f32, row-major [k][n]) into bf16
// transposed layout wts[g][n][k] so WMMA B-fragments are per-lane contiguous.
// ---------------------------------------------------------------------------
__global__ __launch_bounds__(256) void mdrnn_prep_weights(
    const float* __restrict__ W, const float* __restrict__ Ub,
    const float* __restrict__ Uu, const float* __restrict__ Ud,
    __bf16* __restrict__ wts)
{
    int idx = blockIdx.x * 256 + threadIdx.x;   // 0 .. 4*U*U-1
    int g   = idx >> 14;
    int rem = idx & (U_ * U_ - 1);
    int n   = rem >> 7;                         // output dim
    int k   = rem & (U_ - 1);                   // input dim
    const float* src = (g == 0) ? W : (g == 1) ? Ub : (g == 2) ? Uu : Ud;
    wts[idx] = (__bf16)src[k * U_ + n];
}

// hardware tanh (gfx1250 TRANS op); trailing v_nop covers the TRANS data hazard
static __device__ inline float htanh(float v) {
    float r;
    asm("v_tanh_f32 %0, %1\n\tv_nop" : "=v"(r) : "v"(v));
    return r;
}

// ---------------------------------------------------------------------------
// One anti-diagonal wavefront: one block per cell (r, c=d-r).
// 256 threads = 8 wave32; wave w computes output rows [16w, 16w+16) x 128.
// acc += A_g (128x128, bf16) x Wt_g (128x128, bf16) for up to 4 sources.
// All 4 weight matrices staged once via TDM DMA into padded LDS.
// ---------------------------------------------------------------------------
__global__ __launch_bounds__(256) void mdrnn_diag_kernel(
    const float*  __restrict__ x,      // [B][R][C][F] f32
    const float*  __restrict__ bias,   // [U] f32
    const __bf16* __restrict__ wts,    // [4][U][U] bf16 (transposed weights)
    const __bf16* __restrict__ h_d1,   // states of diagonal d-1: [R][B][U]
    const __bf16* __restrict__ h_d2,   // states of diagonal d-2: [R][B][U]
    __bf16*       __restrict__ h_out,  // states of diagonal d:   [R][B][U]
    float*        __restrict__ out,    // final [B][U] (written by last cell)
    int d, int r0)
{
    extern __shared__ __bf16 sW[];     // [4][128][WPAD] = 136 KB dynamic LDS

    const int r      = r0 + (int)blockIdx.x;
    const int c      = d - r;
    const int wave   = threadIdx.x >> 5;
    const int lane   = threadIdx.x & 31;
    const int lane16 = lane & 15;
    const int hi     = lane >> 4;            // 0: lanes 0-15, 1: lanes 16-31

    // A-row for this lane (M index is lane&15 within the wave's 16-row strip)
    const int am = wave * 16 + lane16;
    const float* xrow = x + (((size_t)am * R_ + r) * C_ + c) * F_;
    __builtin_prefetch(xrow, 0, 0);          // global_prefetch_b8 of x slice

    // ---- Stage all 4 weight matrices into padded LDS via the TDM ----
    // 2D tensor: 512 rows x 128 bf16, contiguous in memory (stride 128).
    // LDS padding: rows are 256 B = 64 DWORDs (pad_interval=5), pad 4 DWORDs
    // (pad_amount=3) after each -> 272 B LDS row stride = WPAD bf16 elements.
    if (threadIdx.x < 32) {                  // wave 0 issues the DMA once
        const unsigned long long ga = (unsigned long long)(uintptr_t)wts;
        const unsigned int ldsOff   = (unsigned int)(uintptr_t)sW;
        u32x4 g0;
        g0.x = 1u;                                       // count=1 user D#
        g0.y = ldsOff;                                   // lds_addr (bytes)
        g0.z = (unsigned int)(ga & 0xFFFFFFFFu);         // global_addr[31:0]
        g0.w = (unsigned int)((ga >> 32) & 0x1FFFFFFu)   // global_addr[56:32]
             | (2u << 30);                               // type=2 ("image")
        i32x8 g1;
        g1[0] = (1 << 16)          // data_size = 2 bytes
              | (1 << 20)          // pad_enable
              | (5 << 22)          // pad_interval: 64 DWORDs
              | (3 << 25);         // pad_amount:   4 DWORDs
        g1[1] = (U_ & 0xFFFF) << 16;        // tensor_dim0 = 128 (low 16)
        g1[2] = (4 * U_) << 16;             // tensor_dim1 = 512 (low 16)
        g1[3] = U_ << 16;                   // tile_dim0 = 128
        g1[4] = 4 * U_;                     // tile_dim1 = 512, tile_dim2 = 0
        g1[5] = U_;                         // tensor_dim0_stride = 128
        g1[6] = 0;
        g1[7] = 0;
        const i32x4 z4 = {0, 0, 0, 0};      // groups 2/3 unused (2D tensor)
        const i32x8 z8 = {0, 0, 0, 0, 0, 0, 0, 0};
        __builtin_amdgcn_tensor_load_to_lds(g0, g1, z4, z4, z8, 0);
        __builtin_amdgcn_s_wait_tensorcnt(0);
    }
    __syncthreads();

    v8f acc[8];
#pragma unroll
    for (int nt = 0; nt < 8; ++nt)
        acc[nt] = (v8f){0.f, 0.f, 0.f, 0.f, 0.f, 0.f, 0.f, 0.f};

    // B-frag fetch from padded LDS: K = {k0..k0+7, k0+16..k0+23}, k0 = kc*32 + hi*8
    auto loadB = [&](int g, int kc, int nt) -> v16bf {
        const __bf16* wp =
            &sW[((g << 7) + nt * 16 + lane16) * WPAD + kc * 32 + hi * 8];
        v8bf blo = *(const v8bf*)(wp);
        v8bf bhi = *(const v8bf*)(wp + 16);
        return __builtin_shufflevector(blo, bhi, 0, 1, 2, 3, 4, 5, 6, 7,
                                       8, 9, 10, 11, 12, 13, 14, 15);
    };

#pragma unroll
    for (int g = 0; g < 4; ++g) {
        const bool active =
            (g == 0) ||
            (g == 1 && c > 0) ||
            (g == 2 && r > 0) ||
            (g == 3 && r > 0 && c > 0);
        if (!active) continue;               // block-uniform: EXEC stays full

        const __bf16* hsrc =
            (g == 1) ? h_d1 + (size_t)r * B_ * U_ :          // h(r, c-1)
            (g == 2) ? h_d1 + (size_t)(r - 1) * B_ * U_ :    // h(r-1, c)
                       h_d2 + (size_t)(r - 1) * B_ * U_;     // h(r-1, c-1)

        // ---- Preload all 4 A fragments (16x32 bf16, interleaved K halves) ----
        v16bf a[4];
#pragma unroll
        for (int kc = 0; kc < 4; ++kc) {
            const int k0 = kc * 32 + hi * 8; // {k0..k0+7, k0+16..k0+23}
            if (g == 0) {
                const float* xp = xrow + k0;
                float4 a0 = *(const float4*)(xp + 0);
                float4 a1 = *(const float4*)(xp + 4);
                float4 a2 = *(const float4*)(xp + 16);
                float4 a3 = *(const float4*)(xp + 20);
                v16bf af;
                af[0]  = (__bf16)a0.x; af[1]  = (__bf16)a0.y;
                af[2]  = (__bf16)a0.z; af[3]  = (__bf16)a0.w;
                af[4]  = (__bf16)a1.x; af[5]  = (__bf16)a1.y;
                af[6]  = (__bf16)a1.z; af[7]  = (__bf16)a1.w;
                af[8]  = (__bf16)a2.x; af[9]  = (__bf16)a2.y;
                af[10] = (__bf16)a2.z; af[11] = (__bf16)a2.w;
                af[12] = (__bf16)a3.x; af[13] = (__bf16)a3.y;
                af[14] = (__bf16)a3.z; af[15] = (__bf16)a3.w;
                a[kc] = af;
            } else {
                const __bf16* hp = hsrc + (size_t)am * U_ + k0;
                v8bf lo  = *(const v8bf*)(hp);
                v8bf hi8 = *(const v8bf*)(hp + 16);
                a[kc] = __builtin_shufflevector(lo, hi8, 0, 1, 2, 3, 4, 5, 6, 7,
                                                8, 9, 10, 11, 12, 13, 14, 15);
            }
        }

        // ---- K loop: half-buffer ping-pong so B loads run >=4 WMMAs ahead ----
        v16bf bp[2][4];
#pragma unroll
        for (int j = 0; j < 4; ++j) bp[0][j] = loadB(g, 0, j);
#pragma unroll
        for (int kc = 0; kc < 4; ++kc) {
#pragma unroll
            for (int j = 0; j < 4; ++j) bp[1][j] = loadB(g, kc, 4 + j);
#pragma unroll
            for (int j = 0; j < 4; ++j)
                acc[j] = __builtin_amdgcn_wmma_f32_16x16x32_bf16(
                    false, a[kc], false, bp[0][j], (short)0, acc[j], false, false);
            if (kc < 3) {
#pragma unroll
                for (int j = 0; j < 4; ++j) bp[0][j] = loadB(g, kc + 1, j);
            }
#pragma unroll
            for (int j = 0; j < 4; ++j)
                acc[4 + j] = __builtin_amdgcn_wmma_f32_16x16x32_bf16(
                    false, a[kc], false, bp[1][j], (short)0, acc[4 + j], false, false);
        }
    }

    // ---- Epilogue: bias + hardware tanh, store bf16 state (+ f32 final) ----
    __bf16* ho = h_out + (size_t)r * B_ * U_;
    const bool last = (r == R_ - 1) && (c == C_ - 1);
#pragma unroll
    for (int nt = 0; nt < 8; ++nt) {
        const int n  = nt * 16 + lane16;
        const float bv = bias[n];
#pragma unroll
        for (int e = 0; e < 8; ++e) {
            const int m = wave * 16 + e + hi * 8;   // C/D layout: M = e + 8*hi
            float hval = htanh(acc[nt][e] + bv);
            ho[(size_t)m * U_ + n] = (__bf16)hval;
            if (last) out[(size_t)m * U_ + n] = hval;
        }
    }
}

// ---------------------------------------------------------------------------
// Host launcher
// ---------------------------------------------------------------------------
static inline int imax_(int a, int b) { return a > b ? a : b; }
static inline int imin_(int a, int b) { return a < b ? a : b; }

extern "C" void kernel_launch(void* const* d_in, const int* in_sizes, int n_in,
                              void* d_out, int out_size, void* d_ws, size_t ws_size,
                              hipStream_t stream) {
    const float* x  = (const float*)d_in[0];
    const float* W  = (const float*)d_in[1];
    const float* Ub = (const float*)d_in[2];
    const float* Uu = (const float*)d_in[3];
    const float* Ud = (const float*)d_in[4];
    const float* b  = (const float*)d_in[5];
    float* out = (float*)d_out;

    // Allow the large dynamic LDS allocation (no-op if already set).
    (void)hipFuncSetAttribute((const void*)mdrnn_diag_kernel,
                              hipFuncAttributeMaxDynamicSharedMemorySize,
                              (int)LDS_BYTES);

    // Workspace layout:
    //   [0, 128KB):            4 transposed bf16 weight matrices
    //   then 3 x 2MB:          rotating diagonal state buffers [R][B][U] bf16
    __bf16* wts = (__bf16*)d_ws;
    char* sbase = (char*)d_ws + (size_t)4 * U_ * U_ * sizeof(__bf16);
    const size_t sstride = (size_t)R_ * B_ * U_ * sizeof(__bf16);
    __bf16* st[3] = { (__bf16*)(sbase + 0 * sstride),
                      (__bf16*)(sbase + 1 * sstride),
                      (__bf16*)(sbase + 2 * sstride) };

    mdrnn_prep_weights<<<(4 * U_ * U_) / 256, 256, 0, stream>>>(W, Ub, Uu, Ud, wts);

    for (int d = 0; d < R_ + C_ - 1; ++d) {
        const int r0 = imax_(0, d - (C_ - 1));
        const int r1 = imin_(R_ - 1, d);
        const int ncells = r1 - r0 + 1;
        mdrnn_diag_kernel<<<ncells, 256, LDS_BYTES, stream>>>(
            x, b, wts,
            st[(d + 2) % 3],   // diagonal d-1
            st[(d + 1) % 3],   // diagonal d-2
            st[d % 3],         // diagonal d (write)
            out, d, r0);
    }
}
